// vqa_memnet_90718299226806
// MI455X (gfx1250) — matrix operand order repaired and checked by reading, not compiled, not run on hardware
//
#include <hip/hip_runtime.h>
#include <hip/hip_bf16.h>
#include <math.h>

#define VOCAB 50000
#define Dm    256
#define Em    20000
#define Lm    20

typedef __attribute__((ext_vector_type(16))) _Float16 v16h;
typedef __attribute__((ext_vector_type(8)))  float    v8f;

// ---------- small helpers ----------
__device__ __forceinline__ float4 f4zero() { return make_float4(0.f, 0.f, 0.f, 0.f); }

// acc += row * pe(l, d..d+3), with pe(l,d) = c0 - d*c1
__device__ __forceinline__ void pe_fma4(float4& a, float4 r, float c0, float c1, int d) {
    a.x = fmaf(r.x, c0 - (float)(d + 0) * c1, a.x);
    a.y = fmaf(r.y, c0 - (float)(d + 1) * c1, a.y);
    a.z = fmaf(r.z, c0 - (float)(d + 2) * c1, a.z);
    a.w = fmaf(r.w, c0 - (float)(d + 3) * c1, a.w);
}
__device__ __forceinline__ void add4(float4& a, float4 b) {
    a.x += b.x; a.y += b.y; a.z += b.z; a.w += b.w;
}
__device__ __forceinline__ void axpy4(float4& a, float w, float4 b) {
    a.x = fmaf(w, b.x, a.x); a.y = fmaf(w, b.y, a.y);
    a.z = fmaf(w, b.z, a.z); a.w = fmaf(w, b.w, a.w);
}
__device__ __forceinline__ float dot4(float4 a, float4 b) {
    return a.x * b.x + a.y * b.y + a.z * b.z + a.w * b.w;
}
__device__ __forceinline__ float wave_sum(float v) {
#pragma unroll
    for (int off = 16; off > 0; off >>= 1) v += __shfl_xor(v, off, 32);
    return v;
}
__device__ __forceinline__ void put4h(v16h& v, int o, float4 f) {
    v[o + 0] = (_Float16)f.x; v[o + 1] = (_Float16)f.y;
    v[o + 2] = (_Float16)f.z; v[o + 3] = (_Float16)f.w;
}

// ---------- K0: question embedding (1 block, 256 threads) ----------
__global__ void k_qemb(const int* __restrict__ question,
                       const float* __restrict__ q_table,
                       float* __restrict__ qemb) {
    int d = threadIdx.x;
    float acc = 0.f;
#pragma unroll
    for (int l = 0; l < Lm; ++l) {
        int tok  = question[l];
        float fl = (float)l * (1.0f / Lm);
        float c0 = 1.0f - fl;
        float c1 = (1.0f - 2.0f * fl) * (1.0f / Dm);
        acc = fmaf(q_table[(size_t)tok * Dm + d], c0 - (float)d * c1, acc);
    }
    qemb[d] = acc;
}

// ---------- K1: attention scores, wave-per-evidence (2500 blocks x 256) ----------
__global__ void k_scores(const int* __restrict__ evidence,
                         const float* __restrict__ q_table,
                         const float* __restrict__ te2,
                         const float* __restrict__ qemb,
                         float* __restrict__ scores) {
    int wave = threadIdx.x >> 5;
    int lane = threadIdx.x & 31;
    int e = blockIdx.x * 8 + wave;
    if (e >= Em) return;  // wave-uniform

    const int* ev = evidence + (size_t)e * Lm;
    float4 a0 = f4zero(), a1 = f4zero();
    int d0 = lane * 4, d1 = 128 + lane * 4;
#pragma unroll
    for (int l = 0; l < Lm; ++l) {
        int tok = ev[l];
        const float4* row = (const float4*)(q_table + (size_t)tok * Dm);
        float4 r0 = row[lane];
        float4 r1 = row[32 + lane];
        float fl = (float)l * (1.0f / Lm);
        float c0 = 1.0f - fl;
        float c1 = (1.0f - 2.0f * fl) * (1.0f / Dm);
        pe_fma4(a0, r0, c0, c1, d0);
        pe_fma4(a1, r1, c0, c1, d1);
    }
    const float4* t2 = (const float4*)(te2 + (size_t)e * Dm);
    add4(a0, t2[lane]);
    add4(a1, t2[32 + lane]);

    const float4* qe = (const float4*)qemb;
    float p = dot4(a0, qe[lane]) + dot4(a1, qe[32 + lane]);
    p = wave_sum(p);
    if (lane == 0) scores[e] = p;
}

// ---------- K2/K6: softmax stats (1 block, 1024 threads) ----------
__global__ void k_softmax_stats(const float* __restrict__ x, int n,
                                float* __restrict__ stats) {
    __shared__ float sm[1024];
    int tid = threadIdx.x;
    float m = -INFINITY;
    for (int i = tid; i < n; i += 1024) m = fmaxf(m, x[i]);
    sm[tid] = m; __syncthreads();
    for (int s = 512; s > 0; s >>= 1) {
        if (tid < s) sm[tid] = fmaxf(sm[tid], sm[tid + s]);
        __syncthreads();
    }
    m = sm[0];
    __syncthreads();
    float sum = 0.f;
    for (int i = tid; i < n; i += 1024) sum += expf(x[i] - m);
    sm[tid] = sum; __syncthreads();
    for (int s = 512; s > 0; s >>= 1) {
        if (tid < s) sm[tid] += sm[tid + s];
        __syncthreads();
    }
    if (tid == 0) { stats[0] = m; stats[1] = sm[0]; }
}

// ---------- K3: weighted evidence features, block partials (250 blocks x 256) ----
__global__ void k_weighted(const int* __restrict__ evidence,
                           const float* __restrict__ e_table,
                           const float* __restrict__ te1,
                           const float* __restrict__ scores,
                           const float* __restrict__ stats,
                           float* __restrict__ wpart) {
    __shared__ float lacc[8][Dm];
    int tid  = threadIdx.x;
    int wave = tid >> 5;
    int lane = tid & 31;
    int g = blockIdx.x * 8 + wave;  // 0..1999, 10 evidences each
    float smax = stats[0];
    float inv_ssum = 1.0f / stats[1];

    float4 acc0 = f4zero(), acc1 = f4zero();
    int d0 = lane * 4, d1 = 128 + lane * 4;
    for (int e = g; e < Em; e += 2000) {
        const int* ev = evidence + (size_t)e * Lm;
        float4 f0 = f4zero(), f1 = f4zero();
#pragma unroll
        for (int l = 0; l < Lm; ++l) {
            int tok = ev[l];
            const float4* row = (const float4*)(e_table + (size_t)tok * Dm);
            float4 r0 = row[lane];
            float4 r1 = row[32 + lane];
            float fl = (float)l * (1.0f / Lm);
            float c0 = 1.0f - fl;
            float c1 = (1.0f - 2.0f * fl) * (1.0f / Dm);
            pe_fma4(f0, r0, c0, c1, d0);
            pe_fma4(f1, r1, c0, c1, d1);
        }
        const float4* t1 = (const float4*)(te1 + (size_t)e * Dm);
        add4(f0, t1[lane]);
        add4(f1, t1[32 + lane]);
        float w = expf(scores[e] - smax) * inv_ssum;
        axpy4(acc0, w, f0);
        axpy4(acc1, w, f1);
    }
    // deterministic per-block reduction (disjoint LDS writes, fixed-order sum)
    lacc[wave][d0 + 0] = acc0.x; lacc[wave][d0 + 1] = acc0.y;
    lacc[wave][d0 + 2] = acc0.z; lacc[wave][d0 + 3] = acc0.w;
    lacc[wave][d1 + 0] = acc1.x; lacc[wave][d1 + 1] = acc1.y;
    lacc[wave][d1 + 2] = acc1.z; lacc[wave][d1 + 3] = acc1.w;
    __syncthreads();
    float s = 0.f;
#pragma unroll
    for (int w = 0; w < 8; ++w) s += lacc[w][tid];
    wpart[(size_t)blockIdx.x * Dm + tid] = s;
}

// ---------- K4: features = q_emb + sum of block partials (1 block, 256) --------
__global__ void k_features(const float* __restrict__ qemb,
                           const float* __restrict__ wpart,
                           float* __restrict__ feat) {
    int d = threadIdx.x;
    float s = qemb[d];
    for (int blk = 0; blk < 250; ++blk) s += wpart[(size_t)blk * Dm + d];
    feat[d] = s;
}

// ---------- K5: vocab projection via WMMA (3125 waves, one 16-row tile each) ---
__global__ void k_logits_wmma(const float* __restrict__ W,
                              const float* __restrict__ b,
                              const float* __restrict__ feat,
                              float* __restrict__ logits) {
    int tile = blockIdx.x;       // 16 vocab rows per tile
    int lane = threadIdx.x;      // blockDim = 32, EXEC all ones
    int half = lane >> 4;        // ISA half-wave select for K layout
    int m    = lane & 15;        // A-matrix row within tile
    int row  = tile * 16 + m;

    const float4* wr = (const float4*)(W + (size_t)row * Dm);
    const float4* fr = (const float4*)feat;

    v8f c = {};
#pragma unroll
    for (int it = 0; it < 8; ++it) {     // K = 256 in 8 steps of 32
        // A (16x32 f16): lane<16 holds K {0..7,16..23}; lane>=16 {8..15,24..31}
        int ai = it * 8 + half * 2;
        float4 a0 = wr[ai],     a1 = wr[ai + 1];
        float4 a2 = wr[ai + 4], a3 = wr[ai + 5];
        // B (32x16 f16): lane<16 holds K {0..7,16..23}; lane>=16 {8..15,24..31};
        // broadcast feature chunk to all 16 columns -> all D columns identical
        int bi = it * 8 + half * 2;
        float4 b0 = fr[bi],     b1 = fr[bi + 1];
        float4 b2 = fr[bi + 4], b3 = fr[bi + 5];

        v16h A, B;
        put4h(A, 0, a0);  put4h(A, 4, a1);  put4h(A, 8, a2);  put4h(A, 12, a3);
        put4h(B, 0, b0);  put4h(B, 4, b1);  put4h(B, 8, b2);  put4h(B, 12, b3);

        c = __builtin_amdgcn_wmma_f32_16x16x32_f16(
                /*neg_a=*/false, A, /*neg_b=*/false, B,
                /*c_mod=*/(short)0, c, /*reuse_a=*/false, /*reuse_b=*/false);
    }
    // C/D layout: VGPR r -> M=r (lanes 0..15), M=8+r (lanes 16..31); column N=lane%16.
    // All columns identical, so lanes 0 and 16 carry the 16 results.
    if (m == 0) {
        int base = tile * 16 + half * 8;
#pragma unroll
        for (int r = 0; r < 8; ++r)
            logits[base + r] = c[r] + b[base + r];
    }
}

// ---------- K7: normalized output ----------
__global__ void k_out(const float* __restrict__ logits,
                      const float* __restrict__ stats,
                      float* __restrict__ out) {
    int v = blockIdx.x * 256 + threadIdx.x;
    if (v < VOCAB) out[v] = expf(logits[v] - stats[0]) / stats[1];
}

extern "C" void kernel_launch(void* const* d_in, const int* in_sizes, int n_in,
                              void* d_out, int out_size, void* d_ws, size_t ws_size,
                              hipStream_t stream) {
    (void)in_sizes; (void)n_in; (void)out_size; (void)ws_size;
    const int*   evidence = (const int*)d_in[0];
    const int*   question = (const int*)d_in[1];
    const float* q_table  = (const float*)d_in[2];
    const float* e_table  = (const float*)d_in[3];
    const float* te1      = (const float*)d_in[4];
    const float* te2      = (const float*)d_in[5];
    const float* W        = (const float*)d_in[6];
    const float* b        = (const float*)d_in[7];
    float* out = (float*)d_out;

    float* ws      = (float*)d_ws;
    float* qemb    = ws;                     // 256
    float* feat    = ws + 256;               // 256
    float* scores  = ws + 512;               // 20000
    float* sstats  = ws + 512 + Em;          // 2 (padded to 32)
    float* wpart   = sstats + 32;            // 250*256 = 64000
    float* logits  = wpart + 250 * Dm;       // 50000
    float* lstats  = logits + VOCAB;         // 2

    k_qemb        <<<1,            256,  0, stream>>>(question, q_table, qemb);
    k_scores      <<<Em / 8,       256,  0, stream>>>(evidence, q_table, te2, qemb, scores);
    k_softmax_stats<<<1,           1024, 0, stream>>>(scores, Em, sstats);
    k_weighted    <<<250,          256,  0, stream>>>(evidence, e_table, te1, scores, sstats, wpart);
    k_features    <<<1,            256,  0, stream>>>(qemb, wpart, feat);
    k_logits_wmma <<<VOCAB / 16,   32,   0, stream>>>(W, b, feat, logits);
    k_softmax_stats<<<1,           1024, 0, stream>>>(logits, VOCAB, lstats);
    k_out         <<<(VOCAB + 255) / 256, 256, 0, stream>>>(logits, lstats, out);
}
